// graph_maker2_41343355191811
// MI455X (gfx1250) — matrix operand order repaired and checked by reading, not compiled, not run on hardware
//
#include <hip/hip_runtime.h>
#include <hip/hip_bf16.h>
#include <math.h>

// ---------------------------------------------------------------------------
// Problem constants (from the reference's setup_inputs()).
// ---------------------------------------------------------------------------
#define N_USERS   40000
#define M_ITEMS   20000
#define FEAT      64
#define LATDIM    64
#define KNN       20
#define NNZ_E     1000000
#define NEW_E     (M_ITEMS * KNN)          // 400,000 knn edges (before symmetrize)
#define E_TOT     (NNZ_E + 2 * NEW_E)      // 1,800,000 total edges

// Output layout (d_out, flat, reference return order):
//   [0          , E_TOT)      all_indices row 0   (int32 bits)
//   [E_TOT      , 2*E_TOT)    all_indices row 1   (int32 bits)
//   [2*E_TOT    , 3*E_TOT)    all_values          (all 1.0f)

typedef __attribute__((ext_vector_type(16))) _Float16 v16h;
typedef __attribute__((ext_vector_type(8)))  _Float16 v8h;
typedef __attribute__((ext_vector_type(8)))  float    v8f;

// ---------------------------------------------------------------------------
// Kernel 1: copy original graph indices, set every edge value to 1.0.
// (Reference replaces graph values AND knn values with ones.)
// ---------------------------------------------------------------------------
__global__ __launch_bounds__(256) void fill_graph_kernel(
    const int* __restrict__ gi, int* __restrict__ oIdx, float* __restrict__ oVal)
{
    const int e = blockIdx.x * 256 + threadIdx.x;
    if (e >= E_TOT) return;
    oVal[e] = 1.0f;
    if (e < NNZ_E) {
        oIdx[e]         = gi[e];            // row 0
        oIdx[E_TOT + e] = gi[NNZ_E + e];    // row 1
    }
}

// ---------------------------------------------------------------------------
// Kernel 2: per-item MLP + modal mix + L2 normalize -> en (fp16, row-major
// [M_ITEMS][64]) in workspace. One 64-thread block per item.
// ---------------------------------------------------------------------------
__global__ __launch_bounds__(64) void mlp_norm_kernel(
    const float* __restrict__ xf,   // item_features   [M][64]
    const float* __restrict__ oe,   // original embeds [M][64]
    const float* __restrict__ W0,   // [64][64]
    const float* __restrict__ b0,   // [64]
    const float* __restrict__ W1,   // [64][64]
    const float* __restrict__ b1,   // [64]
    const float* __restrict__ mw,   // modal_weights [2]
    _Float16* __restrict__ en)      // out: normalized embeds fp16 [M][64]
{
    __shared__ float xr[64];
    __shared__ float h0[64];
    __shared__ float sq[64];
    const int row = blockIdx.x;
    const int tid = threadIdx.x;

    xr[tid] = xf[row * FEAT + tid];
    __syncthreads();

    // h0 = relu(x @ W0^T + b0)
    float acc = b0[tid];
    #pragma unroll 8
    for (int f = 0; f < FEAT; ++f) acc += xr[f] * W0[tid * FEAT + f];
    h0[tid] = fmaxf(acc, 0.0f);
    __syncthreads();

    // h1 = h0 @ W1^T + b1
    float acc1 = b1[tid];
    #pragma unroll 8
    for (int f = 0; f < FEAT; ++f) acc1 += h0[f] * W1[tid * FEAT + f];

    // softmax over the 2 modal weights, then mix
    const float e0 = expf(mw[0]);
    const float e1 = expf(mw[1]);
    const float w0 = e0 / (e0 + e1);
    const float emb = w0 * acc1 + (1.0f - w0) * oe[row * LATDIM + tid];

    // L2 norm across the 64 lanes of this row
    sq[tid] = emb * emb;
    __syncthreads();
    for (int s = 32; s > 0; s >>= 1) {
        if (tid < s) sq[tid] += sq[tid + s];
        __syncthreads();
    }
    const float nrm = fmaxf(sqrtf(sq[0]), 1e-12f);
    en[row * LATDIM + tid] = (_Float16)(emb / nrm);
}

// ---------------------------------------------------------------------------
// Kernel 3: fused  sim = en @ en^T  +  per-row top-20  +  edge emission.
//
// Block = 256 threads = 8 waves, owns a 16-row stripe (1250 blocks).
// Per 128-column chunk each wave computes one 16x16 f32 tile with two
// v_wmma_f32_16x16x32_f16 (K=64). Software-pipelined: B fragments for chunk
// i+1 are issued right after chunk i's WMMAs so global latency hides behind
// the scan phase. Double-buffered LDS tile -> one barrier per chunk.
// Top-k scan is wave-parallel: each wave owns 2 rows, 32 lanes test 32
// columns at once and __ballot picks the rare above-threshold candidates.
// ---------------------------------------------------------------------------
__global__ __launch_bounds__(256) void sim_topk_kernel(
    const _Float16* __restrict__ en, int* __restrict__ oIdx)
{
    __shared__ float simT[2][16][128];
    __shared__ float tVal[16][KNN];
    __shared__ int   tIdx[16][KNN];

    const int tid     = threadIdx.x;
    const int lane    = tid & 31;
    const int wv      = tid >> 5;            // wave id 0..7 -> column sub-tile
    const int rowBase = blockIdx.x * 16;

    for (int i = tid; i < 16 * KNN; i += 256) {
        (&tVal[0][0])[i] = -3.0e38f;
        (&tIdx[0][0])[i] = 0;
    }
    __syncthreads();

    // ---- A fragments (rows rowBase..rowBase+15), loop-invariant ----
    // A loaded in contiguous k order (B matches): lane L<16 -> row L, K=0..15; L>=16 -> K=16..31
    const int ar = rowBase + (lane & 15);
    const int ak = (lane & 16) ? 16 : 0;
    union { v16h v; v8h h[2]; } a0, a1;
    a0.h[0] = *(const v8h*)(en + ar * LATDIM + ak);
    a0.h[1] = *(const v8h*)(en + ar * LATDIM + ak + 8);
    a1.h[0] = *(const v8h*)(en + ar * LATDIM + 32 + ak);
    a1.h[1] = *(const v8h*)(en + ar * LATDIM + 32 + ak + 8);

    // ---- prologue: B fragments for chunk 0 ----
    // B loaded in the same contiguous k order as A: lane L<16 -> col L, K=0..15 (32B contiguous en row)
    const int bk  = (lane & 16) ? 16 : 0;
    const int bc0 = wv * 16 + (lane & 15);              // my column within chunk
    v16h bv0 = *(const v16h*)(en + bc0 * LATDIM + bk);
    v16h bv1 = *(const v16h*)(en + bc0 * LATDIM + 32 + bk);

    int buf = 0;
    for (int colBase = 0; colBase < M_ITEMS; colBase += 128, buf ^= 1) {
        const int cidx = colBase + bc0;

        v8f c = {};
        c = __builtin_amdgcn_wmma_f32_16x16x32_f16(false, a0.v, false, bv0,
                                                   (short)0, c, false, false);
        c = __builtin_amdgcn_wmma_f32_16x16x32_f16(false, a1.v, false, bv1,
                                                   (short)0, c, false, false);

        // ---- issue next chunk's B loads now; latency hides behind scan ----
        {
            const int nidx = colBase + 128 + bc0;
            const int nc   = (nidx < M_ITEMS) ? nidx : (M_ITEMS - 1); // clamp
            bv0 = *(const v16h*)(en + nc * LATDIM + bk);
            bv1 = *(const v16h*)(en + nc * LATDIM + 32 + bk);
            // prefetch two chunks ahead into cache (global_prefetch_b8)
            const int pidx = colBase + 256 + bc0;
            const int pc   = (pidx < M_ITEMS) ? pidx : (M_ITEMS - 1);
            __builtin_prefetch(en + pc * LATDIM, 0, 3);
        }

        // ---- scatter tile to LDS, poisoning out-of-range columns ----
        // C/D layout: vgpr i -> M = i (lanes 0-15) / i+8 (lanes 16-31), N = lane&15
        const int  mrow  = (lane < 16) ? 0 : 8;
        const int  ncol  = wv * 16 + (lane & 15);
        const bool valid = (cidx < M_ITEMS);
        #pragma unroll
        for (int i = 0; i < 8; ++i)
            simT[buf][mrow + i][ncol] = valid ? c[i] : -3.0e38f;
        __syncthreads();   // tile visible; also orders scan of buf from 2 chunks ago

        // ---- wave-parallel top-20: each wave owns rows 2*wv, 2*wv+1 ----
        #pragma unroll
        for (int rr = 0; rr < 2; ++rr) {
            const int row = wv * 2 + rr;
            float thresh = tVal[row][KNN - 1];
            #pragma unroll
            for (int step = 0; step < 4; ++step) {
                const float v = simT[buf][row][step * 32 + lane];
                unsigned long long m = __ballot(v > thresh);
                while (m) {
                    const int src = (int)__ffsll(m) - 1;
                    m &= m - 1;
                    const float cv = __shfl(v, src, 32);
                    if (cv > thresh) {
                        // uniform across the wave: all lanes run the identical
                        // insert in lockstep (same addresses, same values)
                        const int cj = colBase + step * 32 + src;
                        int p = KNN - 1;
                        while (p > 0 && tVal[row][p - 1] < cv) {
                            tVal[row][p] = tVal[row][p - 1];
                            tIdx[row][p] = tIdx[row][p - 1];
                            --p;
                        }
                        tVal[row][p] = cv;
                        tIdx[row][p] = cj;
                        thresh = tVal[row][KNN - 1];
                    }
                }
            }
        }
    }
    __syncthreads();

    // ---- emit symmetrized KNN edges (values are ones, written by kernel 1) --
    if (tid < 16) {
        const int r  = rowBase + tid;
        const int ri = r + N_USERS;
        for (int j = 0; j < KNN; ++j) {
            const int e  = r * KNN + j;
            const int ci = tIdx[tid][j] + N_USERS;
            oIdx[NNZ_E + e]                 = ri;   // row 0, forward edge
            oIdx[E_TOT + NNZ_E + e]         = ci;   // row 1, forward edge
            oIdx[NNZ_E + NEW_E + e]         = ci;   // row 0, reverse edge
            oIdx[E_TOT + NNZ_E + NEW_E + e] = ri;   // row 1, reverse edge
        }
    }
}

// ---------------------------------------------------------------------------
// Launch
// ---------------------------------------------------------------------------
extern "C" void kernel_launch(void* const* d_in, const int* in_sizes, int n_in,
                              void* d_out, int out_size, void* d_ws, size_t ws_size,
                              hipStream_t stream) {
    const int*   gi = (const int*)  d_in[0];   // graph_indices [2, NNZ]
    // d_in[1] graph_values: unused (reference overwrites with ones)
    const float* xf = (const float*)d_in[2];   // item_features
    const float* oe = (const float*)d_in[3];   // original_item_embeddings
    const float* W0 = (const float*)d_in[4];
    const float* b0 = (const float*)d_in[5];
    const float* W1 = (const float*)d_in[6];
    const float* b1 = (const float*)d_in[7];
    const float* mw = (const float*)d_in[8];   // modal_weights [2]
    // d_in[9..12]: n_users / m_items / k / b scalars, known constants above.

    _Float16* en   = (_Float16*)d_ws;                       // 2.56 MB fp16
    int*      oIdx = (int*)d_out;                           // indices as int32 bits
    float*    oVal = (float*)d_out + 2 * (size_t)E_TOT;     // values

    fill_graph_kernel<<<(E_TOT + 255) / 256, 256, 0, stream>>>(gi, oIdx, oVal);
    mlp_norm_kernel<<<M_ITEMS, 64, 0, stream>>>(xf, oe, W0, b0, W1, b1, mw, en);
    sim_topk_kernel<<<M_ITEMS / 16, 256, 0, stream>>>(en, oIdx);
}
